// AGCRN_68547678044654
// MI455X (gfx1250) — matrix operand order repaired and checked by reading, not compiled, not run on hardware
//
#include <hip/hip_runtime.h>
#include <cmath>

// ---- problem constants ----
constexpr int kB   = 64;    // batch
constexpr int kT   = 12;    // time steps
constexpr int kN   = 307;   // nodes
constexpr int kNP  = 320;   // nodes padded to WMMA multiple
constexpr int kH   = 64;    // hidden
constexpr int kE   = 10;    // embedding
constexpr int kCH  = 14;    // hyper channels
constexpr int kHM  = 16;    // mask hidden
constexpr int kCIN = 66;    // 2*DIN + H
constexpr int kCP  = 80;    // CIN padded
constexpr int kOG  = 128;   // gate out (2H)
constexpr int kOU  = 64;    // update out (H)
constexpr int kHOR = 12;    // horizon
constexpr int kJ   = 132;   // K*CIN contraction length
constexpr int kJP  = 160;   // padded contraction
constexpr int kNT  = 20;    // node tiles of 16

typedef __attribute__((ext_vector_type(16))) _Float16 v16h;
typedef __attribute__((ext_vector_type(8)))  float    v8f;

union AFrag { v16h v; _Float16 e[16]; unsigned u[8]; };
union CFrag { v8f  v; float    e[8];  };

// K-base for A-fragment dword v (16x32 f16 A layout), lane group grp = lane>>4
__device__ __forceinline__ int aKb(int v, int grp) {
  return (v < 4) ? (2 * v + 8 * grp) : (16 + 2 * (v - 4) + 8 * grp);
}
// K-base for B-fragment dword v (32x16 f16 B layout)
__device__ __forceinline__ int bKb(int v, int grp) { return (v < 4) ? (2 * v + 8 * grp) : (16 + 2 * (v - 4) + 8 * grp); }

__device__ __forceinline__ float sigmoidf(float x) { return 1.f / (1.f + __expf(-x)); }

// ---- precompute: per-node input-adapt weights ----
__global__ void k_weights_h(const float* __restrict__ emb, const float* __restrict__ pool,
                            float* __restrict__ wh) {
  int idx = blockIdx.x * blockDim.x + threadIdx.x;
  if (idx >= kN * kH) return;
  int h = idx % kH, n = idx / kH;
  float s = 0.f;
  for (int d = 0; d < kE; ++d) s += emb[n * kE + d] * pool[d * kH + h];
  wh[idx] = s;
}

// ---- fused hypernet: h_t (never stored) -> x[B,T,N,2], mask projections mp[T,B,N,E] ----
__global__ void k_hyper(const float* __restrict__ hs, const float* __restrict__ hw,
                        const float* __restrict__ hb, const float* __restrict__ wh,
                        const float* __restrict__ src, const float* __restrict__ mW1,
                        const float* __restrict__ mW2, float* __restrict__ x,
                        float* __restrict__ mp) {
  int idx = blockIdx.x * blockDim.x + threadIdx.x;
  if (idx >= kB * kT * kN) return;
  int n = idx % kN; int bt = idx / kN; int t = bt % kT; int b = bt / kT;
  float hv[kH];
#pragma unroll
  for (int h = 0; h < kH; ++h) hv[h] = hb[h];
  for (int c = 0; c < kCH; ++c) {
    float s = hs[((b * kCH + c) * kT + t) * kN + n];
#pragma unroll
    for (int h = 0; h < kH; ++h) hv[h] += s * hw[c * kH + h];
  }
#pragma unroll
  for (int h = 0; h < kH; ++h) hv[h] = tanhf(hv[h]);
  float xa = 0.f;
#pragma unroll
  for (int h = 0; h < kH; ++h) xa += hv[h] * wh[n * kH + h];
  x[((b * kT + t) * kN + n) * 2 + 0] = xa;
  x[((b * kT + t) * kN + n) * 2 + 1] = src[(b * kT + t) * kN + n];
  float tmp[kHM];
  for (int hm = 0; hm < kHM; ++hm) {
    float s = 0.f;
#pragma unroll
    for (int h = 0; h < kH; ++h) s += hv[h] * mW1[h * kHM + hm];
    tmp[hm] = s;
  }
  for (int e = 0; e < kE; ++e) {
    float s = 0.f;
#pragma unroll
    for (int hm = 0; hm < kHM; ++hm) s += tmp[hm] * mW2[hm * kE + e];
    mp[((size_t)t * kB + b) * kN * kE + (size_t)n * kE + e] = s;
  }
}

// ---- support A = 0.5*(row-normalized adj + softmax(relu(E E^T))) ----
__global__ void k_support(const float* __restrict__ adj, const float* __restrict__ emb,
                          float* __restrict__ A) {
  const int n = blockIdx.x, tid = threadIdx.x;
  __shared__ float sv[kN];
  __shared__ float red[128];
  float en[kE];
  for (int e = 0; e < kE; ++e) en[e] = emb[n * kE + e];
  float lmax = 0.f, asum = 0.f;
  for (int m = tid; m < kN; m += 128) {
    float d = 0.f;
    for (int e = 0; e < kE; ++e) d += en[e] * emb[m * kE + e];
    d = fmaxf(d, 0.f);
    sv[m] = d;
    lmax = fmaxf(lmax, d);
    asum += adj[n * kN + m];
  }
  red[tid] = lmax; __syncthreads();
  for (int s = 64; s > 0; s >>= 1) { if (tid < s) red[tid] = fmaxf(red[tid], red[tid + s]); __syncthreads(); }
  float gmax = red[0]; __syncthreads();
  red[tid] = asum; __syncthreads();
  for (int s = 64; s > 0; s >>= 1) { if (tid < s) red[tid] += red[tid + s]; __syncthreads(); }
  float adjsum = red[0]; __syncthreads();
  float epart = 0.f;
  for (int m = tid; m < kN; m += 128) { float e2 = __expf(sv[m] - gmax); sv[m] = e2; epart += e2; }
  red[tid] = epart; __syncthreads();
  for (int s = 64; s > 0; s >>= 1) { if (tid < s) red[tid] += red[tid + s]; __syncthreads(); }
  float esum = red[0]; __syncthreads();
  float ia = 1.f / fmaxf(adjsum, 1e-6f), ie = 1.f / esum;
  for (int m = tid; m < kN; m += 128)
    A[n * kN + m] = 0.5f * (adj[n * kN + m] * ia + sv[m] * ie);
}

// ---- per-node graph-conv weights, f16, B-fragment layout [N][O][JP], zero-padded ----
__global__ void k_genW(const float* __restrict__ emb, const float* __restrict__ pool,
                       _Float16* __restrict__ Wt, int O, int total) {
  int idx = blockIdx.x * blockDim.x + threadIdx.x;
  if (idx >= total) return;
  int j = idx % kJP; int rem = idx / kJP; int o = rem % O; int n = rem / O;
  float s = 0.f;
  if (j < kJ) {
    int k = j / kCIN, i = j % kCIN;
    for (int d = 0; d < kE; ++d) s += emb[n * kE + d] * pool[((d * 2 + k) * kCIN + i) * O + o];
  }
  Wt[idx] = (_Float16)s;
}
__global__ void k_genB(const float* __restrict__ emb, const float* __restrict__ bpool,
                       float* __restrict__ bout, int O, int total) {
  int idx = blockIdx.x * blockDim.x + threadIdx.x;
  if (idx >= total) return;
  int o = idx % O; int n = idx / O;
  float s = 0.f;
  for (int d = 0; d < kE; ++d) s += emb[n * kE + d] * bpool[d * O + o];
  bout[idx] = s;
}

// ---- per-(t,b): masked support mA[b][320][320] f16, diag sigmoid mdiag[b][320] ----
__global__ void k_mask(const float* __restrict__ mpt, const float* __restrict__ emb,
                       const float* __restrict__ A, _Float16* __restrict__ mA,
                       float* __restrict__ mdiag) {
  int idx = blockIdx.x * 256 + threadIdx.x;
  const int b = blockIdx.y;
  if (idx >= kNP * kNP) return;
  int n = idx / kNP, m = idx % kNP;
  float val = 0.f, sg = 0.f;
  if (n < kN && m < kN) {
    float d = 0.f;
    for (int e = 0; e < kE; ++e) d += mpt[(size_t)(b * kN + n) * kE + e] * emb[m * kE + e];
    sg = sigmoidf(d);
    val = A[n * kN + m] * sg;
  }
  mA[((size_t)b * kNP + n) * kNP + m] = (_Float16)val;
  if (m == n) mdiag[b * kNP + n] = sg;
}

// ---- per-step: build transposed f16 xs_t[b][80][320]; seed cand_t (x rows + zeros) ----
__global__ void k_build(const float* __restrict__ x, const float* __restrict__ state,
                        _Float16* __restrict__ xs_t, _Float16* __restrict__ cand_t, int t) {
  int idx = blockIdx.x * blockDim.x + threadIdx.x;
  if (idx >= kB * kCP * kNP) return;
  int m = idx % kNP; int rem = idx / kNP; int i = rem % kCP; int b = rem / kCP;
  float val = 0.f;
  if (m < kN && i < kCIN)
    val = (i < 2) ? x[((b * kT + t) * kN + m) * 2 + i]
                  : state[(size_t)(b * kN + m) * kH + (i - 2)];
  xs_t[idx] = (_Float16)val;
  cand_t[idx] = (i < 2) ? (_Float16)val : (_Float16)0.f;
}

// ---- masked graph aggregation (WMMA, no LDS): per (b, 16-node tile); 5 waves = 5 col tiles
//      xg[n][b][66+i] = (A.*mask) @ v ; xg[n][b][i] = diag-mask * v  (f16, [N][64][160]) ----
__global__ __launch_bounds__(160) void k_agg(const _Float16* __restrict__ vin,   // [B][80][320]
                                             const _Float16* __restrict__ mA,    // [B][320][320]
                                             const float* __restrict__ mdiag,    // [B][320]
                                             _Float16* __restrict__ xg) {        // [N][64][160]
  const int b = blockIdx.y, nbase = blockIdx.x * 16;
  const int tid = threadIdx.x, lane = tid & 31, w = tid >> 5;
  const int grp = lane >> 4, lr = lane & 15;
  const _Float16* arow = mA + ((size_t)b * kNP + (nbase + lr)) * kNP;
  const int i = w * 16 + lr;
  const _Float16* brow = vin + ((size_t)b * kCP + i) * kNP;
  CFrag acc;
#pragma unroll
  for (int q = 0; q < 8; ++q) acc.e[q] = 0.f;

  for (int kc = 0; kc < 10; ++kc) {          // K = 320
    AFrag a, bf;
#pragma unroll
    for (int v = 0; v < 8; ++v) a.u[v] = *(const unsigned*)(arow + kc * 32 + aKb(v, grp));
#pragma unroll
    for (int v = 0; v < 8; ++v) bf.u[v] = *(const unsigned*)(brow + kc * 32 + bKb(v, grp));
    acc.v = __builtin_amdgcn_wmma_f32_16x16x32_f16(false, a.v, false, bf.v,
                                                   (short)0, acc.v, false, false);
  }
#pragma unroll
  for (int vv = 0; vv < 8; ++vv) {           // k=1 half of xg (j = 66+i)
    int n = nbase + vv + (grp << 3);
    if (n < kN && i < kCIN)
      xg[((size_t)n * kB + b) * kJP + kCIN + i] = (_Float16)acc.e[vv];
  }
  for (int idx = tid; idx < 16 * kCIN; idx += 160) {   // k=0 half (j = i): diag scale
    int r = idx / kCIN, i0 = idx % kCIN, n = nbase + r;
    if (n < kN) {
      float vv = (float)vin[((size_t)b * kCP + i0) * kNP + n];
      xg[((size_t)n * kB + b) * kJP + i0] = (_Float16)(mdiag[b * kNP + n] * vv);
    }
  }
}

// ---- gate apply (WMMA): zr = sigmoid(xg @ Wg[n] + bg); z*state -> cand_t, r -> rbuf ----
__global__ __launch_bounds__(256) void k_gate(const _Float16* __restrict__ xg,  // [N][64][160]
                                              const _Float16* __restrict__ Wg,  // [N][128][160]
                                              const float* __restrict__ bg,
                                              const float* __restrict__ state,
                                              _Float16* __restrict__ cand_t,    // [B][80][320]
                                              float* __restrict__ rbuf) {
  const int n = blockIdx.x;
  const int tid = threadIdx.x, lane = tid & 31, w = tid >> 5;
  const int mt = w >> 1, hsel = w & 1;
  const int grp = lane >> 4, lr = lane & 15;
  const _Float16* abase = xg + ((size_t)n * kB + (mt * 16 + lr)) * kJP;
  CFrag acc[4];
#pragma unroll
  for (int c = 0; c < 4; ++c)
#pragma unroll
    for (int q = 0; q < 8; ++q) acc[c].e[q] = 0.f;
  for (int kc = 0; kc < 5; ++kc) {           // K = 160
    AFrag a;
#pragma unroll
    for (int v = 0; v < 8; ++v) a.u[v] = *(const unsigned*)(abase + kc * 32 + aKb(v, grp));
#pragma unroll
    for (int ct = 0; ct < 4; ++ct) {
      int o = (hsel * 4 + ct) * 16 + lr;
      const _Float16* bbase = Wg + ((size_t)n * kOG + o) * kJP + kc * 32;
      AFrag bf;
#pragma unroll
      for (int v = 0; v < 8; ++v) bf.u[v] = *(const unsigned*)(bbase + bKb(v, grp));
      acc[ct].v = __builtin_amdgcn_wmma_f32_16x16x32_f16(false, a.v, false, bf.v,
                                                         (short)0, acc[ct].v, false, false);
    }
  }
#pragma unroll
  for (int ct = 0; ct < 4; ++ct) {
    int o = (hsel * 4 + ct) * 16 + lr;
    float bias = bg[n * kOG + o];
#pragma unroll
    for (int vv = 0; vv < 8; ++vv) {
      int bb = mt * 16 + vv + (grp << 3);
      float s = sigmoidf(acc[ct].e[vv] + bias);
      if (o < kH)
        cand_t[((size_t)bb * kCP + (2 + o)) * kNP + n] =
            (_Float16)(s * state[(size_t)(bb * kN + n) * kH + o]);
      else
        rbuf[(size_t)(bb * kN + n) * kH + (o - kH)] = s;
    }
  }
}

// ---- update apply (WMMA): hc = tanh(xg2 @ Wu[n] + bu); state = r*state + (1-r)*hc ----
__global__ __launch_bounds__(256) void k_upd(const _Float16* __restrict__ xg,   // [N][64][160]
                                             const _Float16* __restrict__ Wu,   // [N][64][160]
                                             const float* __restrict__ bu,
                                             const float* __restrict__ rbuf,
                                             float* __restrict__ state) {
  const int n = blockIdx.x;
  const int tid = threadIdx.x, lane = tid & 31, w = tid >> 5;
  const int mt = w >> 1, hsel = w & 1;
  const int grp = lane >> 4, lr = lane & 15;
  const _Float16* abase = xg + ((size_t)n * kB + (mt * 16 + lr)) * kJP;
  CFrag acc[2];
#pragma unroll
  for (int c = 0; c < 2; ++c)
#pragma unroll
    for (int q = 0; q < 8; ++q) acc[c].e[q] = 0.f;
  for (int kc = 0; kc < 5; ++kc) {
    AFrag a;
#pragma unroll
    for (int v = 0; v < 8; ++v) a.u[v] = *(const unsigned*)(abase + kc * 32 + aKb(v, grp));
#pragma unroll
    for (int ct = 0; ct < 2; ++ct) {
      int o = (hsel * 2 + ct) * 16 + lr;
      const _Float16* bbase = Wu + ((size_t)n * kOU + o) * kJP + kc * 32;
      AFrag bf;
#pragma unroll
      for (int v = 0; v < 8; ++v) bf.u[v] = *(const unsigned*)(bbase + bKb(v, grp));
      acc[ct].v = __builtin_amdgcn_wmma_f32_16x16x32_f16(false, a.v, false, bf.v,
                                                         (short)0, acc[ct].v, false, false);
    }
  }
#pragma unroll
  for (int ct = 0; ct < 2; ++ct) {
    int o = (hsel * 2 + ct) * 16 + lr;
    float bias = bu[n * kOU + o];
#pragma unroll
    for (int vv = 0; vv < 8; ++vv) {
      int bb = mt * 16 + vv + (grp << 3);
      size_t idx = (size_t)(bb * kN + n) * kH + o;
      float hc = tanhf(acc[ct].e[vv] + bias);
      float r = rbuf[idx], st = state[idx];
      state[idx] = r * st + (1.f - r) * hc;
    }
  }
}

// ---- layernorm + end conv ----
__global__ void k_final(const float* __restrict__ state, const float* __restrict__ gam,
                        const float* __restrict__ bet, const float* __restrict__ eW,
                        const float* __restrict__ eb, float* __restrict__ out) {
  int bn = blockIdx.x; int b = bn / kN, n = bn % kN; int h = threadIdx.x;
  __shared__ float sh[kH];
  __shared__ float sx[kH];
  float s = state[(size_t)bn * kH + h];
  sh[h] = s; __syncthreads();
  float mu = 0.f;
#pragma unroll
  for (int i = 0; i < kH; ++i) mu += sh[i];
  mu *= (1.f / kH);
  float var = 0.f;
#pragma unroll
  for (int i = 0; i < kH; ++i) { float d = sh[i] - mu; var += d * d; }
  var *= (1.f / kH);
  float xn = (s - mu) * rsqrtf(var + 1e-12f) * gam[h] + bet[h];
  sx[h] = xn; __syncthreads();
  if (h < kHOR) {
    float a = eb[h];
#pragma unroll
    for (int i = 0; i < kH; ++i) a += sx[i] * eW[h * kH + i];
    out[((size_t)b * kHOR + h) * kN + n] = a;
  }
}

extern "C" void kernel_launch(void* const* d_in, const int* in_sizes, int n_in,
                              void* d_out, int out_size, void* d_ws, size_t ws_size,
                              hipStream_t stream) {
  const float* hs  = (const float*)d_in[0];
  const float* src = (const float*)d_in[1];
  const float* adj = (const float*)d_in[2];
  const float* emb = (const float*)d_in[3];
  const float* pool= (const float*)d_in[4];
  const float* hw  = (const float*)d_in[5];
  const float* hb  = (const float*)d_in[6];
  const float* mW1 = (const float*)d_in[7];
  const float* mW2 = (const float*)d_in[8];
  const float* gWp = (const float*)d_in[9];
  const float* gbp = (const float*)d_in[10];
  const float* uWp = (const float*)d_in[11];
  const float* ubp = (const float*)d_in[12];
  const float* gam = (const float*)d_in[13];
  const float* bet = (const float*)d_in[14];
  const float* eW  = (const float*)d_in[15];
  const float* eb  = (const float*)d_in[16];
  float* out = (float*)d_out;

  char* ws = (char*)d_ws;
  size_t off = 0;
  auto alloc = [&](size_t bytes) -> char* {
    char* p = ws + off;
    off = (off + bytes + 255) & ~(size_t)255;
    return p;
  };
  float*    wh    = (float*)alloc((size_t)kN * kH * 4);
  float*    x     = (float*)alloc((size_t)kB * kT * kN * 2 * 4);
  float*    mp    = (float*)alloc((size_t)kT * kB * kN * kE * 4);
  float*    A     = (float*)alloc((size_t)kN * kN * 4);
  _Float16* Wg    = (_Float16*)alloc((size_t)kN * kOG * kJP * 2);
  _Float16* Wu    = (_Float16*)alloc((size_t)kN * kOU * kJP * 2);
  float*    bg    = (float*)alloc((size_t)kN * kOG * 4);
  float*    bu    = (float*)alloc((size_t)kN * kOU * 4);
  float*    state = (float*)alloc((size_t)kB * kN * kH * 4);
  _Float16* xs_t  = (_Float16*)alloc((size_t)kB * kCP * kNP * 2);
  _Float16* cand_t= (_Float16*)alloc((size_t)kB * kCP * kNP * 2);
  _Float16* xg    = (_Float16*)alloc((size_t)kN * kB * kJP * 2);
  _Float16* mA    = (_Float16*)alloc((size_t)kB * kNP * kNP * 2);
  float*    mdiag = (float*)alloc((size_t)kB * kNP * 4);
  float*    rbuf  = (float*)alloc((size_t)kB * kN * kH * 4);

  // t-invariant precompute
  k_weights_h<<<(kN * kH + 255) / 256, 256, 0, stream>>>(emb, pool, wh);
  k_hyper<<<(kB * kT * kN + 127) / 128, 128, 0, stream>>>(hs, hw, hb, wh, src, mW1, mW2, x, mp);
  k_support<<<kN, 128, 0, stream>>>(adj, emb, A);
  { int tot = kN * kOG * kJP; k_genW<<<(tot + 255) / 256, 256, 0, stream>>>(emb, gWp, Wg, kOG, tot); }
  { int tot = kN * kOU * kJP; k_genW<<<(tot + 255) / 256, 256, 0, stream>>>(emb, uWp, Wu, kOU, tot); }
  { int tot = kN * kOG; k_genB<<<(tot + 255) / 256, 256, 0, stream>>>(emb, gbp, bg, kOG, tot); }
  { int tot = kN * kOU; k_genB<<<(tot + 255) / 256, 256, 0, stream>>>(emb, ubp, bu, kOU, tot); }
  hipMemsetAsync(state, 0, (size_t)kB * kN * kH * 4, stream);
  hipMemsetAsync(xg, 0, (size_t)kN * kB * kJP * 2, stream);   // j in [132,160) stays zero

  // recurrent scan
  for (int t = 0; t < kT; ++t) {
    const float* mpt = mp + (size_t)t * kB * kN * kE;
    k_mask<<<dim3((kNP * kNP + 255) / 256, kB), 256, 0, stream>>>(mpt, emb, A, mA, mdiag);
    k_build<<<(kB * kCP * kNP + 255) / 256, 256, 0, stream>>>(x, state, xs_t, cand_t, t);
    k_agg<<<dim3(kNT, kB), 160, 0, stream>>>(xs_t, mA, mdiag, xg);
    k_gate<<<kN, 256, 0, stream>>>(xg, Wg, bg, state, cand_t, rbuf);
    k_agg<<<dim3(kNT, kB), 160, 0, stream>>>(cand_t, mA, mdiag, xg);
    k_upd<<<kN, 256, 0, stream>>>(xg, Wu, bu, rbuf, state);
  }
  k_final<<<kB * kN, kH, 0, stream>>>(state, gam, bet, eW, eb, out);
}
